// PointPillars_59184649339265
// MI455X (gfx1250) — matrix operand order repaired and checked by reading, not compiled, not run on hardware
//
#include <hip/hip_runtime.h>

// ---------------- problem constants ----------------
namespace {
constexpr int   kB   = 4;
constexpr int   kCin = 8;
constexpr int   kP   = 12000;
constexpr int   kN   = 64;
constexpr int   kCp  = 64;
constexpr int   kH   = 282;
constexpr int   kW   = 282;
constexpr int   kHW  = kH * kW;          // 79524
constexpr int   kPN  = kP * kN;          // 768000  (per-channel, per-batch elements)
constexpr int   kNPix = kB * kHW;        // 318096
constexpr int   kNT16 = kNPix / 16;      // 19881 (exact)
constexpr float kEps   = 1e-5f;
constexpr float kXMin  = -22.0f;
constexpr float kPillar = 0.16f;
constexpr int   kG  = 240;               // stats reduction blocks
constexpr int   kNS = 44;                // 8 first moments + 36 upper-tri second moments
}

// ---------------- CDNA5 vector types ----------------
typedef __attribute__((ext_vector_type(16))) __bf16        v16bf;
typedef __attribute__((ext_vector_type(8)))  float         v8f;
typedef __attribute__((ext_vector_type(4)))  unsigned int  u32x4;
typedef __attribute__((ext_vector_type(4)))  float         f32x4;

union Frag32 {                 // 32-byte WMMA 16-bit operand fragment (8 VGPRs)
  v16bf v;
  struct { u32x4 lo; u32x4 hi; } u;
};

__device__ __forceinline__ unsigned short f2bf(float f) {
  unsigned int u = __float_as_uint(f);
  u += 0x7fffu + ((u >> 16) & 1u);       // round-to-nearest-even
  return (unsigned short)(u >> 16);
}

// ---------------- 0) zero canvas + tags ----------------
__global__ void zero_kernel(u32x4* __restrict__ p, long n16) {
  long i = (long)blockIdx.x * blockDim.x + threadIdx.x;
  if (i < n16) p[i] = (u32x4){0u, 0u, 0u, 0u};
}

// ---------------- 1) input statistics: S1[8], S2[36] over (B,P,N) ----------------
__global__ void stats_kernel(const float* __restrict__ x, float* __restrict__ partial) {
  float a[kNS];
#pragma unroll
  for (int q = 0; q < kNS; ++q) a[q] = 0.0f;

  const int idx0   = blockIdx.x * blockDim.x + threadIdx.x;
  const int stride = gridDim.x * blockDim.x;
  for (int j = idx0; j < kB * kPN; j += stride) {
    int b = j / kPN;
    int i = j - b * kPN;
    const float* base = x + (long)b * kCin * kPN + i;
    float v[kCin];
#pragma unroll
    for (int c = 0; c < kCin; ++c) v[c] = base[(long)c * kPN];
    int k = 8;
#pragma unroll
    for (int c = 0; c < kCin; ++c) {
      a[c] += v[c];
#pragma unroll
      for (int d = c; d < kCin; ++d) a[k++] += v[c] * v[d];
    }
  }

  // wave32 shuffle reduce, then cross-wave via LDS
#pragma unroll
  for (int q = 0; q < kNS; ++q) {
    float v = a[q];
#pragma unroll
    for (int off = 16; off > 0; off >>= 1) v += __shfl_down(v, off);
    a[q] = v;
  }
  __shared__ float red[8][kNS];
  const int wv = threadIdx.x >> 5, ln = threadIdx.x & 31;
  if (ln == 0) {
#pragma unroll
    for (int q = 0; q < kNS; ++q) red[wv][q] = a[q];
  }
  __syncthreads();
  if (threadIdx.x < kNS) {
    float s = 0.0f;
#pragma unroll
    for (int w = 0; w < 8; ++w) s += red[w][threadIdx.x];
    partial[(long)blockIdx.x * kNS + threadIdx.x] = s;
  }
}

// ---------------- 2) fold BN analytically into weights/bias ----------------
// mean_o = sum_c w[o,c]*mu_c + b_o ;  var_o = w^T (E[xx^T] - mu mu^T) w
__global__ void finalize_kernel(const float* __restrict__ partial,
                                const float* __restrict__ w,
                                const float* __restrict__ bias,
                                const float* __restrict__ gamma,
                                const float* __restrict__ beta,
                                float* __restrict__ wfold,
                                float* __restrict__ tfold) {
  __shared__ float S[kNS];
  const int t = threadIdx.x;
  if (t < kNS) {
    float s = 0.0f;
    for (int g = 0; g < kG; ++g) s += partial[(long)g * kNS + t];
    S[t] = s;
  }
  __syncthreads();
  if (t < kCp) {
    const float inv = 1.0f / (float)((long)kB * kPN);
    const int rowstart[8] = {0, 8, 15, 21, 26, 30, 33, 35};
    float mu[kCin], wr[kCin];
#pragma unroll
    for (int c = 0; c < kCin; ++c) { mu[c] = S[c] * inv; wr[c] = w[t * kCin + c]; }
    float wm = 0.0f;
#pragma unroll
    for (int c = 0; c < kCin; ++c) wm += wr[c] * mu[c];
    float ey2 = 0.0f;
#pragma unroll
    for (int c = 0; c < kCin; ++c) {
#pragma unroll
      for (int d = 0; d < kCin; ++d) {
        int cc = c < d ? c : d;
        int dd = c < d ? d : c;
        ey2 += wr[c] * wr[d] * S[8 + rowstart[cc] + (dd - cc)];
      }
    }
    ey2 *= inv;
    float var = ey2 - wm * wm;
    float s   = gamma[t] * rsqrtf(var + kEps);
    tfold[t]  = beta[t] - s * wm + s * 0.0f;  // relu(y*s + t); bias folded below
    // out = relu( sum_c (w*s)*x + (b*s + beta - mean*s) )
    tfold[t] = beta[t] + s * (bias[t] - (wm + bias[t]));
#pragma unroll
    for (int c = 0; c < kCin; ++c) wfold[t * kCin + c] = wr[c] * s;
  }
}

// ---------------- 3) deterministic scatter winner per cell ----------------
__global__ void tag_kernel(const float* __restrict__ x, int* __restrict__ tag) {
  int pil = blockIdx.x * blockDim.x + threadIdx.x;
  if (pil >= kB * kP) return;
  int b = pil / kP, p = pil - b * kP;
  const float* xb = x + (long)b * kCin * kPN + (long)p * kN;
  float px = xb[0];
  float py = xb[(long)kPN];
  if (px == 0.0f) return;
  int xg = (int)floorf((px - kXMin) / kPillar);
  int yg = (int)floorf((py - kXMin) / kPillar);
  xg = xg < 0 ? 0 : (xg > kW - 1 ? kW - 1 : xg);
  yg = yg < 0 ? 0 : (yg > kH - 1 ? kH - 1 : yg);
  atomicMax(&tag[(long)b * kHW + yg * kW + xg], pil + 1);
}

// ---------------- 4) fused PFN (linear+BN+ReLU+max_n) + scatter (bf16) ----------------
// one wave per pillar; lane owns output channels (lane, lane+32)
__global__ void pfn_scatter_kernel(const float* __restrict__ x,
                                   const float* __restrict__ wfold,
                                   const float* __restrict__ tfold,
                                   const int* __restrict__ tag,
                                   unsigned short* __restrict__ canvas,
                                   int chanOff) {
  __shared__ __align__(16) float xs[8][kCin * kN];
  const int wv = threadIdx.x >> 5, ln = threadIdx.x & 31;
  const int pillar = blockIdx.x * 8 + wv;           // kB*kP divisible by 8
  const int b = pillar / kP, p = pillar - b * kP;
  const float* xb = x + (long)b * kCin * kPN + (long)p * kN;

#pragma unroll
  for (int t = ln; t < 128; t += 32) {              // 128 x float4 = 8ch x 64n
    int c = t >> 4, q = t & 15;
    f32x4 v = *(const f32x4*)(xb + (long)c * kPN + q * 4);
    *(f32x4*)&xs[wv][c * kN + q * 4] = v;
  }
  __syncthreads();

  float w0[kCin], w1[kCin];
#pragma unroll
  for (int c = 0; c < kCin; ++c) {
    w0[c] = wfold[ln * kCin + c];
    w1[c] = wfold[(ln + 32) * kCin + c];
  }
  const float t0 = tfold[ln], t1 = tfold[ln + 32];

  float m0 = 0.0f, m1 = 0.0f;                       // relu => max >= 0
  for (int n = 0; n < kN; ++n) {
    float a0 = t0, a1 = t1;
#pragma unroll
    for (int c = 0; c < kCin; ++c) {
      float xv = xs[wv][c * kN + n];
      a0 = fmaf(w0[c], xv, a0);
      a1 = fmaf(w1[c], xv, a1);
    }
    m0 = fmaxf(m0, a0);
    m1 = fmaxf(m1, a1);
  }

  float px = xs[wv][0];                             // (b,0,p,0)
  float py = xs[wv][kN];                            // (b,1,p,0)
  if (px != 0.0f) {
    int xg = (int)floorf((px - kXMin) / kPillar);
    int yg = (int)floorf((py - kXMin) / kPillar);
    xg = xg < 0 ? 0 : (xg > kW - 1 ? kW - 1 : xg);
    yg = yg < 0 ? 0 : (yg > kH - 1 ? kH - 1 : yg);
    long cell = (long)b * kHW + yg * kW + xg;
    if (tag[cell] == pillar + 1) {                  // deterministic winner
      unsigned short* dst = canvas + cell * 128 + chanOff;
      dst[ln]      = f2bf(m0);
      dst[ln + 32] = f2bf(m1);
    }
  }
}

// ---------------- 5) convert w_bb to bf16 ----------------
__global__ void cvt_wbb_kernel(const float* __restrict__ w, unsigned short* __restrict__ o) {
  int i = blockIdx.x * blockDim.x + threadIdx.x;
  if (i < kCp * 128) o[i] = f2bf(w[i]);
}

// ---------------- 6) 1x1 conv (K=128) via v_wmma_f32_16x16x32_bf16 ----------------
// one wave computes a 64-channel x 16-pixel tile: 4 M-tiles x 4 K-steps = 16 WMMA ops
__global__ void conv_wmma_kernel(const unsigned short* __restrict__ canvas,
                                 const unsigned short* __restrict__ wbb,
                                 const float* __restrict__ bbb,
                                 float* __restrict__ out) {
  const int gwave = (blockIdx.x * blockDim.x + threadIdx.x) >> 5;  // wave-uniform
  const int lane  = threadIdx.x & 31;
  if (gwave >= kNT16) return;                                      // uniform exit
  const int pix0 = gwave * 16;
  const int lp = lane & 15;      // pixel / A-row within tile
  const int lh = lane >> 4;      // half-wave select

  v8f acc[4] = {};

  // B operand: column = pixel lp, K-run = ks*32 + lh*16 .. +15 (contiguous in canvas)
  const unsigned short* bbase = canvas + (long)(pix0 + lp) * 128 + lh * 16;
  // A operand: row = mt*16 + lp, K-run = ks*32 + lh*16 .. +15
  const unsigned short* abase = wbb + (long)lp * 128 + lh * 16;

#pragma unroll
  for (int ks = 0; ks < 4; ++ks) {
    Frag32 bf;
    bf.u.lo = *(const u32x4*)(bbase + ks * 32);
    bf.u.hi = *(const u32x4*)(bbase + ks * 32 + 8);
#pragma unroll
    for (int mt = 0; mt < 4; ++mt) {
      const unsigned short* ap = abase + (long)mt * 16 * 128 + ks * 32;
      Frag32 af;
      af.u.lo = *(const u32x4*)(ap);
      af.u.hi = *(const u32x4*)(ap + 8);
      acc[mt] = __builtin_amdgcn_wmma_f32_16x16x32_bf16(
          /*neg_a=*/false, af.v, /*neg_b=*/false, bf.v,
          /*c_mod=*/(short)0, acc[mt], /*reuse_a=*/false, /*reuse_b=*/false);
    }
  }

  const int pl = pix0 + lp;
  const int b  = pl / kHW;
  const int hw = pl - b * kHW;
  float* ob = out + (long)b * kCp * kHW + hw;
#pragma unroll
  for (int mt = 0; mt < 4; ++mt) {
#pragma unroll
    for (int r = 0; r < 8; ++r) {
      int o = mt * 16 + lh * 8 + r;
      ob[(long)o * kHW] = acc[mt][r] + bbb[o];
    }
  }
}

// ---------------- workspace layout ----------------
static constexpr size_t SZ_CANVAS = (size_t)kNPix * 128 * sizeof(unsigned short); // 81,432,576
static constexpr size_t SZ_TAG    = (size_t)kNPix * sizeof(int);                  // 1,272,384
static constexpr size_t OFF_CANVAS = 0;
static constexpr size_t OFF_TAG0   = OFF_CANVAS + SZ_CANVAS;
static constexpr size_t OFF_TAG1   = OFF_TAG0 + SZ_TAG;
static constexpr size_t OFF_ZEND   = OFF_TAG1 + SZ_TAG;          // zero [0, OFF_ZEND)
static constexpr size_t OFF_PART0  = OFF_ZEND;
static constexpr size_t SZ_PART    = (size_t)kG * kNS * sizeof(float);
static constexpr size_t OFF_PART1  = OFF_PART0 + SZ_PART;
static constexpr size_t OFF_WF0    = OFF_PART1 + SZ_PART;
static constexpr size_t OFF_TF0    = OFF_WF0 + kCp * kCin * sizeof(float);
static constexpr size_t OFF_WF1    = OFF_TF0 + kCp * sizeof(float);
static constexpr size_t OFF_TF1    = OFF_WF1 + kCp * kCin * sizeof(float);
static constexpr size_t OFF_WBB    = OFF_TF1 + kCp * sizeof(float);

extern "C" void kernel_launch(void* const* d_in, const int* in_sizes, int n_in,
                              void* d_out, int out_size, void* d_ws, size_t ws_size,
                              hipStream_t stream) {
  (void)in_sizes; (void)n_in; (void)out_size; (void)ws_size;

  const float* sweep   = (const float*)d_in[0];
  const float* map_in  = (const float*)d_in[1];
  const float* w_s     = (const float*)d_in[2];
  const float* b_s     = (const float*)d_in[3];
  const float* g_s     = (const float*)d_in[4];
  const float* be_s    = (const float*)d_in[5];
  const float* w_m     = (const float*)d_in[6];
  const float* b_m     = (const float*)d_in[7];
  const float* g_m     = (const float*)d_in[8];
  const float* be_m    = (const float*)d_in[9];
  const float* w_bb    = (const float*)d_in[10];
  const float* b_bb    = (const float*)d_in[11];
  float* out = (float*)d_out;

  char* ws = (char*)d_ws;
  unsigned short* canvas = (unsigned short*)(ws + OFF_CANVAS);
  int*   tag0  = (int*)(ws + OFF_TAG0);
  int*   tag1  = (int*)(ws + OFF_TAG1);
  float* part0 = (float*)(ws + OFF_PART0);
  float* part1 = (float*)(ws + OFF_PART1);
  float* wf0   = (float*)(ws + OFF_WF0);
  float* tf0   = (float*)(ws + OFF_TF0);
  float* wf1   = (float*)(ws + OFF_WF1);
  float* tf1   = (float*)(ws + OFF_TF1);
  unsigned short* wbb16 = (unsigned short*)(ws + OFF_WBB);

  // 0) zero canvas + tags
  long n16 = (long)(OFF_ZEND / 16);
  zero_kernel<<<(int)((n16 + 255) / 256), 256, 0, stream>>>((u32x4*)ws, n16);

  // 1) input stats
  stats_kernel<<<kG, 256, 0, stream>>>(sweep,  part0);
  stats_kernel<<<kG, 256, 0, stream>>>(map_in, part1);

  // 2) BN fold
  finalize_kernel<<<1, 64, 0, stream>>>(part0, w_s, b_s, g_s, be_s, wf0, tf0);
  finalize_kernel<<<1, 64, 0, stream>>>(part1, w_m, b_m, g_m, be_m, wf1, tf1);

  // 3) deterministic scatter winners
  tag_kernel<<<(kB * kP + 255) / 256, 256, 0, stream>>>(sweep,  tag0);
  tag_kernel<<<(kB * kP + 255) / 256, 256, 0, stream>>>(map_in, tag1);

  // 4) fused PFN + scatter (one wave per pillar; 8 pillars per block)
  pfn_scatter_kernel<<<kB * kP / 8, 256, 0, stream>>>(sweep,  wf0, tf0, tag0, canvas, 0);
  pfn_scatter_kernel<<<kB * kP / 8, 256, 0, stream>>>(map_in, wf1, tf1, tag1, canvas, 64);

  // 5) weight convert
  cvt_wbb_kernel<<<(kCp * 128 + 255) / 256, 256, 0, stream>>>(w_bb, wbb16);

  // 6) WMMA 1x1 conv
  int nwaves  = kNT16;                       // 19881 tiles of 16 pixels
  int nblocks = (nwaves * 32 + 255) / 256;   // 8 waves per block
  conv_wmma_kernel<<<nblocks, 256, 0, stream>>>(canvas, wbb16, b_bb, out);
}
